// Net2_80676665688183
// MI455X (gfx1250) — compile-verified, matrix operand order checked by reading, not run on hardware
//
#include <hip/hip_runtime.h>
#include <hip/hip_bf16.h>
#include <math.h>

#define HEADS 3
#define CCH   32
#define D1    96          // HEADS*CCH (layer widths, both layers)
#define NEG_SLOPE 0.2f
#define EPS_DEN   1e-16f

typedef __bf16 bf16_t;
typedef bf16_t v16bf __attribute__((ext_vector_type(16)));
typedef float  v8f   __attribute__((ext_vector_type(8)));

// fragment K-index for element i of a 16-bit WMMA operand (ISA 7.12.2)
__device__ __forceinline__ int frag_kk(int i, int khalf)
{
    const int v = i >> 1, s = i & 1;
    return ((v >> 2) << 4) + (khalf << 3) + ((v & 3) << 1) + s;
}

// ---------------------------------------------------------------------------
// Pack W[K,Ncol] (f32, row-major) into fragment-major bf16:
// pw[((chunk*NT + tile)*32 + lane)*16 + i] = bf16(W[(chunk*32 + kk(lane,i))*Ncol
//                                                   + tile*16 + (lane&15)])
// One thread per (chunk, tile, lane). W is tiny and massively reused.
// ---------------------------------------------------------------------------
__global__ void pack_w_kernel(const float* __restrict__ W, bf16_t* __restrict__ pw,
                              int K, int Ncol)
{
    const int NT  = Ncol >> 4;
    const int tot = (K >> 5) * NT * 32;
    const int idx = blockIdx.x * blockDim.x + threadIdx.x;
    if (idx >= tot) return;
    const int l     = idx & 31;
    const int t     = (idx >> 5) % NT;
    const int c     = idx / (32 * NT);
    const int khalf = l >> 4, mlo = l & 15;
#pragma unroll
    for (int i = 0; i < 16; ++i) {
        const int kk = frag_kk(i, khalf);
        pw[(long long)idx * 16 + i] =
            (bf16_t)W[(long long)(c * 32 + kk) * Ncol + t * 16 + mlo];
    }
}

// ---------------------------------------------------------------------------
// WMMA GEMM: Y[M,NCOL] = X[M,K] @ W + bias ; act: 0=none, 1=relu
// - B operand comes from pre-packed bf16 `pw`, staged into LDS with
//   global_load_async_to_lds_b128 (ASYNCcnt) once per block.
// - A operand: four global_load_b128 per k-chunk (two contiguous 8-float runs).
// - CT column tiles per wave share one A fragment -> CT back-to-back v_wmma.
// 4 waves/block (one 16-row stripe each). M%16==0, K%32==0, NCOL%(CT*16)==0.
// ---------------------------------------------------------------------------
template <int CT, int K, int NCOL>
__global__ __launch_bounds__(128)
void gemm_bias_wmma(const float* __restrict__ X, const bf16_t* __restrict__ pw,
                    const float* __restrict__ bias, float* __restrict__ Y,
                    int M, int act)
{
    constexpr int NCHUNK = K / 32;
    constexpr int NT     = NCOL / 16;
    __shared__ bf16_t sW[NCHUNK * CT * 512];     // [chunk][ct][lane][16]

    const int lane  = threadIdx.x & 31;
    const int wave  = threadIdx.x >> 5;
    const int mlo   = lane & 15;
    const int khalf = lane >> 4;
    const int t0    = blockIdx.y * CT;           // first column tile of this block

    // ---- stage this block's packed-W stripe into LDS (async DMA path) ----
    constexpr int UNITS = NCHUNK * CT * 64;      // 16-byte units
    for (int u = threadIdx.x; u < UNITS; u += 128) {
        const int c = u / (CT * 64);
        const int r = u % (CT * 64);
        const bf16_t* gp = pw + ((long long)(c * NT + t0) * 512 + r * 8);
        const unsigned lds = (unsigned)(size_t)&sW[c * CT * 512 + r * 8];
        asm volatile("global_load_async_to_lds_b128 %0, %1, off"
                     :: "v"(lds), "v"(gp) : "memory");
    }
    asm volatile("s_wait_asynccnt 0" ::: "memory");
    __syncthreads();

    const int rowTile = blockIdx.x * 4 + wave;
    if (rowTile * 16 < M) {                      // wave-uniform; after barrier
        const int rowBase = rowTile * 16;
        const float* xrow = X + (long long)(rowBase + mlo) * K;

        v8f acc[CT];
#pragma unroll
        for (int c = 0; c < CT; ++c) acc[c] = (v8f){};

#pragma unroll
        for (int k0 = 0; k0 < K; k0 += 32) {
            // A fragment: two contiguous 8-float runs -> 4x b128
            const float4 f0 = *(const float4*)(xrow + k0 + khalf * 8);
            const float4 f1 = *(const float4*)(xrow + k0 + khalf * 8 + 4);
            const float4 f2 = *(const float4*)(xrow + k0 + 16 + khalf * 8);
            const float4 f3 = *(const float4*)(xrow + k0 + 16 + khalf * 8 + 4);
            v16bf a;
            a[0]  = (bf16_t)f0.x; a[1]  = (bf16_t)f0.y;
            a[2]  = (bf16_t)f0.z; a[3]  = (bf16_t)f0.w;
            a[4]  = (bf16_t)f1.x; a[5]  = (bf16_t)f1.y;
            a[6]  = (bf16_t)f1.z; a[7]  = (bf16_t)f1.w;
            a[8]  = (bf16_t)f2.x; a[9]  = (bf16_t)f2.y;
            a[10] = (bf16_t)f2.z; a[11] = (bf16_t)f2.w;
            a[12] = (bf16_t)f3.x; a[13] = (bf16_t)f3.y;
            a[14] = (bf16_t)f3.z; a[15] = (bf16_t)f3.w;

            const int cc = k0 >> 5;
#pragma unroll
            for (int c = 0; c < CT; ++c) {
                const v16bf b = *(const v16bf*)&sW[((cc * CT + c) << 9) + lane * 16];
                // (neg_a, A, neg_b, B, c_mod, C, reuse_a, reuse_b)
                acc[c] = __builtin_amdgcn_wmma_f32_16x16x32_bf16(
                             false, a, false, b, (short)0, acc[c], false, false);
            }
        }

#pragma unroll
        for (int c = 0; c < CT; ++c) {
            const int col  = (t0 + c) * 16 + mlo;
            const float bc = bias[col];
#pragma unroll
            for (int v = 0; v < 8; ++v) {
                const int r = rowBase + v + 8 * khalf;   // M % 16 == 0
                float val = acc[c][v] + bc;
                if (act == 1) val = fmaxf(val, 0.0f);
                Y[(long long)r * NCOL + col] = val;
            }
        }
    }
}

// ---------------------------------------------------------------------------
// Init per-node softmax state: m = -inf, denom = 0, acc = 0
// ---------------------------------------------------------------------------
__global__ void init_state_kernel(float* __restrict__ m, float* __restrict__ denom,
                                  float* __restrict__ acc, int N)
{
    const int i = blockIdx.x * blockDim.x + threadIdx.x;
    if (i < N * HEADS) { m[i] = -INFINITY; denom[i] = 0.0f; }
    if (i < N * D1)    acc[i] = 0.0f;
}

// int/uint punned float atomic max (handles any sign mix; m initialized -inf)
__device__ inline void atomicMaxFloat(float* addr, float val)
{
    if (!(val < 0.0f)) atomicMax((int*)addr, __float_as_int(val));
    else               atomicMin((unsigned int*)addr, __float_as_uint(val));
}

// ---------------------------------------------------------------------------
// Pass 1: per-edge attention logits + per-destination segment max.
// One wave per edge; lane l owns channel h*32+l for each head h (C==wave32).
// ---------------------------------------------------------------------------
__global__ void edge_logits_max_kernel(const int* __restrict__ ei, int E, int N,
                                       const float* __restrict__ xl,
                                       const float* __restrict__ xr,
                                       const float* __restrict__ att,
                                       float* __restrict__ logits,
                                       float* __restrict__ m)
{
    const int wid  = (blockIdx.x * blockDim.x + threadIdx.x) >> 5;
    const int lane = threadIdx.x & 31;
    const int Etot = E + N;
    if (wid >= Etot) return;

    const int s = (wid < E) ? ei[wid]     : (wid - E);   // src (self-loop tail)
    const int d = (wid < E) ? ei[E + wid] : (wid - E);   // dst

    __builtin_prefetch(xl + (long long)s * D1, 0, 0);
    __builtin_prefetch(xr + (long long)d * D1, 0, 0);

    float part[HEADS];
#pragma unroll
    for (int h = 0; h < HEADS; ++h) {
        const int idx = h * CCH + lane;
        const float v = xl[(long long)s * D1 + idx] + xr[(long long)d * D1 + idx];
        const float e = (v > 0.0f) ? v : NEG_SLOPE * v;   // leaky_relu
        part[h] = e * att[idx];
    }
#pragma unroll
    for (int off = 16; off > 0; off >>= 1) {
#pragma unroll
        for (int h = 0; h < HEADS; ++h) part[h] += __shfl_xor(part[h], off, 32);
    }
    if (lane == 0) {
#pragma unroll
        for (int h = 0; h < HEADS; ++h) {
            logits[(long long)wid * HEADS + h] = part[h];
            atomicMaxFloat(&m[d * HEADS + h], part[h]);
        }
    }
}

// ---------------------------------------------------------------------------
// Pass 2: a = exp(logit - m[dst]); denom[dst] += a   (a overwrites logits)
// ---------------------------------------------------------------------------
__global__ void edge_exp_sum_kernel(const int* __restrict__ ei, int E, int N,
                                    const float* __restrict__ m,
                                    float* __restrict__ a,
                                    float* __restrict__ denom)
{
    const int e = blockIdx.x * blockDim.x + threadIdx.x;
    const int Etot = E + N;
    if (e >= Etot) return;
    const int d = (e < E) ? ei[E + e] : (e - E);
#pragma unroll
    for (int h = 0; h < HEADS; ++h) {
        const float av = __expf(a[(long long)e * HEADS + h] - m[d * HEADS + h]);
        a[(long long)e * HEADS + h] = av;
        atomicAdd(&denom[d * HEADS + h], av);
    }
}

// ---------------------------------------------------------------------------
// Pass 3: acc[dst] += x_l[src] * alpha  (alpha = a/(denom+eps)); wave/edge.
// ---------------------------------------------------------------------------
__global__ void edge_scatter_kernel(const int* __restrict__ ei, int E, int N,
                                    const float* __restrict__ xl,
                                    const float* __restrict__ a,
                                    const float* __restrict__ denom,
                                    float* __restrict__ acc)
{
    const int wid  = (blockIdx.x * blockDim.x + threadIdx.x) >> 5;
    const int lane = threadIdx.x & 31;
    const int Etot = E + N;
    if (wid >= Etot) return;

    const int s = (wid < E) ? ei[wid]     : (wid - E);
    const int d = (wid < E) ? ei[E + wid] : (wid - E);

    __builtin_prefetch(xl + (long long)s * D1, 0, 0);

#pragma unroll
    for (int h = 0; h < HEADS; ++h) {
        const float alpha = a[(long long)wid * HEADS + h] /
                            (denom[d * HEADS + h] + EPS_DEN);
        const int idx = h * CCH + lane;
        atomicAdd(&acc[(long long)d * D1 + idx],
                  xl[(long long)s * D1 + idx] * alpha);
    }
}

// ---------------------------------------------------------------------------
// h = tanh(acc + conv_bias)
// ---------------------------------------------------------------------------
__global__ void node_finish_kernel(const float* __restrict__ acc,
                                   const float* __restrict__ bias,
                                   float* __restrict__ h, int N)
{
    const long long i = (long long)blockIdx.x * blockDim.x + threadIdx.x;
    if (i >= (long long)N * D1) return;
    h[i] = tanhf(acc[i] + bias[i % D1]);
}

// ---------------------------------------------------------------------------
// first_idx[g] = searchsorted(batch, g)  (batch sorted, all graphs non-empty)
// ---------------------------------------------------------------------------
__global__ void first_idx_kernel(const int* __restrict__ batch, int N, int G,
                                 int* __restrict__ fidx)
{
    const int g = blockIdx.x * blockDim.x + threadIdx.x;
    if (g >= G) return;
    int lo = 0, hi = N;
    while (lo < hi) { int mid = (lo + hi) >> 1; if (batch[mid] < g) lo = mid + 1; else hi = mid; }
    fidx[g] = lo;
}

// pooled[g] = concat(h1[first_idx[g]], h2[first_idx[g]])  -> [G, 192]
__global__ void pool_kernel(const int* __restrict__ fidx,
                            const float* __restrict__ h1,
                            const float* __restrict__ h2,
                            float* __restrict__ pooled, int G)
{
    const int t = blockIdx.x * blockDim.x + threadIdx.x;
    if (t >= G * 2 * D1) return;
    const int g = t / (2 * D1);
    const int j = t % (2 * D1);
    const int i = fidx[g];
    pooled[t] = (j < D1) ? h1[(long long)i * D1 + j]
                         : h2[(long long)i * D1 + (j - D1)];
}

// logits = hidden @ W_fc2 + b_fc2 ; out = log_softmax (2 classes, stable)
__global__ void head_kernel(const float* __restrict__ hidden,
                            const float* __restrict__ W2,
                            const float* __restrict__ b2,
                            float* __restrict__ out, int G, int H)
{
    const int g = blockIdx.x * blockDim.x + threadIdx.x;
    if (g >= G) return;
    float z0 = b2[0], z1 = b2[1];
    for (int k = 0; k < H; ++k) {
        const float hv = hidden[(long long)g * H + k];
        z0 = fmaf(hv, W2[k * 2 + 0], z0);
        z1 = fmaf(hv, W2[k * 2 + 1], z1);
    }
    const float mx  = fmaxf(z0, z1);
    const float lse = mx + logf(expf(z0 - mx) + expf(z1 - mx));
    out[g * 2 + 0] = z0 - lse;
    out[g * 2 + 1] = z1 - lse;
}

// ---------------------------------------------------------------------------
extern "C" void kernel_launch(void* const* d_in, const int* in_sizes, int n_in,
                              void* d_out, int out_size, void* d_ws, size_t ws_size,
                              hipStream_t stream)
{
    const float* x      = (const float*)d_in[0];
    const float* W_l1   = (const float*)d_in[1];
    const float* b_l1   = (const float*)d_in[2];
    const float* W_r1   = (const float*)d_in[3];
    const float* b_r1   = (const float*)d_in[4];
    const float* att1   = (const float*)d_in[5];
    const float* bias1  = (const float*)d_in[6];
    const float* W_l2   = (const float*)d_in[7];
    const float* b_l2   = (const float*)d_in[8];
    const float* W_r2   = (const float*)d_in[9];
    const float* b_r2   = (const float*)d_in[10];
    const float* att2   = (const float*)d_in[11];
    const float* bias2  = (const float*)d_in[12];
    const float* W_fc1  = (const float*)d_in[13];
    const float* b_fc1  = (const float*)d_in[14];
    const float* W_fc2  = (const float*)d_in[15];
    const float* b_fc2  = (const float*)d_in[16];
    const int*   ei     = (const int*)d_in[17];
    const int*   batch  = (const int*)d_in[18];

    const int IN_DIM = 128;
    const int HIDDEN = 512;
    const int N    = in_sizes[0] / IN_DIM;
    const int E    = in_sizes[17] / 2;
    const int Etot = E + N;
    const int G    = out_size / 2;

    // ---- workspace carve-out (floats) ----
    float* ws     = (float*)d_ws;
    float* xl     = ws;                              // [N, 96]
    float* xr     = xl     + (long long)N * D1;      // [N, 96]
    float* acc    = xr     + (long long)N * D1;      // [N, 96]
    float* h1     = acc    + (long long)N * D1;      // [N, 96]
    float* h2     = h1     + (long long)N * D1;      // [N, 96]
    float* alog   = h2     + (long long)N * D1;      // [Etot, 3]  logits -> a
    float* m      = alog   + (long long)Etot * HEADS;// [N, 3]
    float* denom  = m      + (long long)N * HEADS;   // [N, 3]
    float* pooled = denom  + (long long)N * HEADS;   // [G, 192]
    float* hidden = pooled + (long long)G * 2 * D1;  // [G, 512]
    int*   fidx   = (int*)(hidden + (long long)G * HIDDEN); // [G]
    bf16_t* pw    = (bf16_t*)(fidx + G);             // packed W, <= 98304 bf16

    const int rowTiles = (N + 15) / 16;              // 3125
    const dim3 gemmBlk(128);                         // 4 waves, 1 row tile each
    const dim3 gemmGridN((rowTiles + 3) / 4, 1);     // NCOL=96 -> CT=6, grid.y=1
    const int edgeWaveGrid = (Etot * 32 + 255) / 256;
    const int edgeThrGrid  = (Etot + 255) / 256;
    const int nodeGrid     = ((long long)N * D1 + 255) / 256;

    const int packN1 = (IN_DIM / 32) * (D1 / 16) * 32;       // K=128 packs
    const int packN2 = (D1 / 32) * (D1 / 16) * 32;           // K=96  packs
    const int packNF = ((2 * D1) / 32) * (HIDDEN / 16) * 32; // K=192 packs

    // ================= Layer 1 (K = 128) =================
    pack_w_kernel<<<(packN1 + 255) / 256, 256, 0, stream>>>(W_l1, pw, IN_DIM, D1);
    gemm_bias_wmma<6, 128, 96><<<gemmGridN, gemmBlk, 0, stream>>>(x, pw, b_l1, xl, N, 0);
    pack_w_kernel<<<(packN1 + 255) / 256, 256, 0, stream>>>(W_r1, pw, IN_DIM, D1);
    gemm_bias_wmma<6, 128, 96><<<gemmGridN, gemmBlk, 0, stream>>>(x, pw, b_r1, xr, N, 0);
    init_state_kernel<<<nodeGrid, 256, 0, stream>>>(m, denom, acc, N);
    edge_logits_max_kernel<<<edgeWaveGrid, 256, 0, stream>>>(ei, E, N, xl, xr, att1, alog, m);
    edge_exp_sum_kernel  <<<edgeThrGrid, 256, 0, stream>>>(ei, E, N, m, alog, denom);
    edge_scatter_kernel  <<<edgeWaveGrid, 256, 0, stream>>>(ei, E, N, xl, alog, denom, acc);
    node_finish_kernel   <<<nodeGrid, 256, 0, stream>>>(acc, bias1, h1, N);

    // ================= Layer 2 (K = 96) =================
    pack_w_kernel<<<(packN2 + 255) / 256, 256, 0, stream>>>(W_l2, pw, D1, D1);
    gemm_bias_wmma<6, 96, 96><<<gemmGridN, gemmBlk, 0, stream>>>(h1, pw, b_l2, xl, N, 0);
    pack_w_kernel<<<(packN2 + 255) / 256, 256, 0, stream>>>(W_r2, pw, D1, D1);
    gemm_bias_wmma<6, 96, 96><<<gemmGridN, gemmBlk, 0, stream>>>(h1, pw, b_r2, xr, N, 0);
    init_state_kernel<<<nodeGrid, 256, 0, stream>>>(m, denom, acc, N);
    edge_logits_max_kernel<<<edgeWaveGrid, 256, 0, stream>>>(ei, E, N, xl, xr, att2, alog, m);
    edge_exp_sum_kernel  <<<edgeThrGrid, 256, 0, stream>>>(ei, E, N, m, alog, denom);
    edge_scatter_kernel  <<<edgeWaveGrid, 256, 0, stream>>>(ei, E, N, xl, alog, denom, acc);
    node_finish_kernel   <<<nodeGrid, 256, 0, stream>>>(acc, bias2, h2, N);

    // ================= Readout + MLP =================
    first_idx_kernel<<<(G + 255) / 256, 256, 0, stream>>>(batch, N, G, fidx);
    pool_kernel<<<(G * 2 * D1 + 255) / 256, 256, 0, stream>>>(fidx, h1, h2, pooled, G);
    pack_w_kernel<<<(packNF + 255) / 256, 256, 0, stream>>>(W_fc1, pw, 2 * D1, HIDDEN);
    {
        // 512x192x512 GEMM, fused ReLU: CT=4 -> grid.y = 512/(4*16) = 8
        dim3 grid((G / 16 + 3) / 4, HIDDEN / (4 * 16));
        gemm_bias_wmma<4, 192, 512><<<grid, gemmBlk, 0, stream>>>(pooled, pw, b_fc1,
                                                                  hidden, G, 1);
    }
    head_kernel<<<(G + 255) / 256, 256, 0, stream>>>(hidden, W_fc2, b_fc2,
                                                     (float*)d_out, G, HIDDEN);
}
